// Decoder_28999619183233
// MI455X (gfx1250) — matrix operand order repaired and checked by reading, not compiled, not run on hardware
//
#include <hip/hip_runtime.h>
#include <stdint.h>

// ---------------------------------------------------------------------------
// Types for CDNA5 WMMA (gfx1250, wave32)
// ---------------------------------------------------------------------------
typedef __attribute__((ext_vector_type(16))) _Float16 v16h;
typedef __attribute__((ext_vector_type(8)))  _Float16 v8h;
typedef __attribute__((ext_vector_type(8)))  float    v8f;

#define B_   32
#define T_   256
#define D_   512
#define H_   8
#define DK_  64
#define FF_  2048
#define TKM_ 1024   // encoder memory length
#define MASKNEG (-30000.0f)

__device__ __forceinline__ v8h ld8(const _Float16* p) { return *(const v8h*)p; }

// A-matrix fragment (16x32 f16, MxK): lane holds row M=lane%16,
// K chunks [8*(lane>>4) .. +7] and [+16 .. +23]  (ISA 7.12.2)
__device__ __forceinline__ v16h load_a_frag(const _Float16* base, int stride, int k0) {
    int lane = threadIdx.x & 31;
    const _Float16* p = base + (long)(lane & 15) * stride + k0 + ((lane >> 4) << 3);
    union { v16h v; v8h h[2]; } f;
    f.h[0] = ld8(p);
    f.h[1] = ld8(p + 16);
    return f.v;
}

// B-matrix fragment (32x16 f16, KxN) loaded from Bt stored N-major (N x K):
// lane holds col N=lane%16, K chunks [8*(lane>>4) .. +7] and [+16 .. +23]
__device__ __forceinline__ v16h load_b_frag(const _Float16* base, int stride, int k0) {
    int lane = threadIdx.x & 31;
    const _Float16* p = base + (long)(lane & 15) * stride + k0 + ((lane >> 4) << 3);
    union { v16h v; v8h h[2]; } f;
    f.h[0] = ld8(p);
    f.h[1] = ld8(p + 16);
    return f.v;
}

__device__ __forceinline__ v8f wmma_f16(v16h a, v16h b, v8f c) {
    return __builtin_amdgcn_wmma_f32_16x16x32_f16(
        /*neg_a=*/false, a, /*neg_b=*/false, b,
        /*c_mod=*/(short)0, c, /*reuse_a=*/false, /*reuse_b=*/false);
}

// Async global -> LDS 16B copy (ASYNCcnt-tracked, ISA 15.18.3).
// Inline asm: portable across both toolchains (builtin arity differs).
__device__ __forceinline__ void async_copy16(const _Float16* gptr, const _Float16* lptr) {
    unsigned lds_addr = (unsigned)(unsigned long long)(const void*)lptr;
    unsigned long long gaddr = (unsigned long long)(const void*)gptr;
    asm volatile("global_load_async_to_lds_b128 %0, %1, off"
                 :: "v"(lds_addr), "v"(gaddr) : "memory");
}
__device__ __forceinline__ void wait_async0() {
    asm volatile("s_wait_asynccnt 0x0" ::: "memory");
}

// ---------------------------------------------------------------------------
// Embedding + positional encoding:  X[b,t,:] = emb[tok]*sqrt(D) + PE(t,:)
// ---------------------------------------------------------------------------
__global__ __launch_bounds__(256) void embed_pe_kernel(
    const int* __restrict__ target, const float* __restrict__ emb,
    float* __restrict__ X) {
    int bt  = blockIdx.x;
    int t   = bt & (T_ - 1);
    int tok = target[bt];
    for (int c = threadIdx.x; c < D_; c += 256) {
        int   i2  = c & ~1;
        float div = __expf((float)i2 * (-9.210340371976184f / (float)D_));
        float ang = (float)t * div;
        float pe  = (c & 1) ? __cosf(ang) : __sinf(ang);
        X[(long)bt * D_ + c] = emb[(long)tok * D_ + c] * 22.627416997969522f + pe;
    }
}

// ---------------------------------------------------------------------------
// Weight convert + transpose: W (KxN f32) -> Wt (NxK f16)
// ---------------------------------------------------------------------------
__global__ __launch_bounds__(256) void wt_convert_kernel(
    const float* __restrict__ W, _Float16* __restrict__ Wt, int K, int N) {
    long i = (long)blockIdx.x * 256 + threadIdx.x;
    if (i < (long)K * N) {
        long k = i / N, n = i % N;
        Wt[n * (long)K + k] = (_Float16)W[i];
    }
}

__global__ __launch_bounds__(256) void f2h_kernel(
    const float* __restrict__ src, _Float16* __restrict__ dst, long n) {
    long i = (long)blockIdx.x * 256 + threadIdx.x;
    if (i < n) dst[i] = (_Float16)src[i];
}

// ---------------------------------------------------------------------------
// LayerNorm over D=512, one row per block (256 threads, 2 elements/thread)
// ---------------------------------------------------------------------------
__global__ __launch_bounds__(256) void ln_kernel(
    const float* __restrict__ X, const float* __restrict__ g,
    const float* __restrict__ bt, _Float16* __restrict__ o16,
    float* __restrict__ o32) {
    __shared__ float red[256];
    long row = blockIdx.x;
    int  tid = threadIdx.x;
    float x0 = X[row * D_ + tid];
    float x1 = X[row * D_ + tid + 256];

    red[tid] = x0 + x1;
    __syncthreads();
    for (int s = 128; s > 0; s >>= 1) {
        if (tid < s) red[tid] += red[tid + s];
        __syncthreads();
    }
    float mean = red[0] * (1.0f / D_);
    __syncthreads();

    float d0 = x0 - mean, d1 = x1 - mean;
    red[tid] = d0 * d0 + d1 * d1;
    __syncthreads();
    for (int s = 128; s > 0; s >>= 1) {
        if (tid < s) red[tid] += red[tid + s];
        __syncthreads();
    }
    float rstd = rsqrtf(red[0] * (1.0f / D_) + 1e-6f);

    float y0 = d0 * rstd * g[tid]       + bt[tid];
    float y1 = d1 * rstd * g[tid + 256] + bt[tid + 256];
    if (o16) { o16[row * D_ + tid] = (_Float16)y0; o16[row * D_ + tid + 256] = (_Float16)y1; }
    if (o32) { o32[row * D_ + tid] = y0;           o32[row * D_ + tid + 256] = y1; }
}

// ---------------------------------------------------------------------------
// f16 WMMA GEMM:  C[M,N] = A[M,K] @ Wt[N,K]^T + bias
// Block tile 128x128, 8 waves (4M x 2N), wave tile 32x64, K-step 32.
// A/B tiles staged through LDS with double-buffered async global->LDS copies.
// MODE: 0 = f16 out          1 = f16 out, V-transposed (B,H,DK,VTT)
//       2 = f32 out + res    3 = f16 out + ReLU
// ---------------------------------------------------------------------------
template <int MODE, int VTT>
__global__ __launch_bounds__(256) void gemm_f16_kernel(
    const _Float16* __restrict__ A, const _Float16* __restrict__ Wt,
    const float* __restrict__ bias, const float* residual,
    float* out32, _Float16* out16, _Float16* outVt,
    int M, int N, int K) {
    __shared__ __align__(16) _Float16 Abuf[2][128 * 32];
    __shared__ __align__(16) _Float16 Bbuf[2][128 * 32];

    int tid = threadIdx.x, lane = tid & 31, wid = tid >> 5;
    int wm = wid & 3, wn = wid >> 2;
    long mB = (long)blockIdx.y * 128;
    long nB = (long)blockIdx.x * 128;

    // Issue async copies for one K-slice: 512 16B-chunks per tile, 2/thread.
    auto issue = [&](int buf, int k0) {
#pragma unroll
        for (int j = 0; j < 2; ++j) {
            int c   = tid + j * 256;        // chunk id
            int row = c >> 2;               // 4 chunks per 32-elem row
            int kc  = (c & 3) << 3;         // element offset within row
            async_copy16(A  + (mB + row) * K + k0 + kc, &Abuf[buf][row * 32 + kc]);
            async_copy16(Wt + (nB + row) * K + k0 + kc, &Bbuf[buf][row * 32 + kc]);
        }
    };

    v8f acc[2][4];
#pragma unroll
    for (int mt = 0; mt < 2; ++mt)
#pragma unroll
        for (int nt = 0; nt < 4; ++nt) acc[mt][nt] = (v8f){};

    issue(0, 0);
    int cur = 0;
    for (int k0 = 0; k0 < K; k0 += 32) {
        wait_async0();
        __syncthreads();                 // tile[cur] ready; tile[cur^1] drained
        if (k0 + 32 < K) issue(cur ^ 1, k0 + 32);

        v16h a[2], b[4];
#pragma unroll
        for (int mt = 0; mt < 2; ++mt)
            a[mt] = load_a_frag(&Abuf[cur][(wm * 32 + mt * 16) * 32], 32, 0);
#pragma unroll
        for (int nt = 0; nt < 4; ++nt)
            b[nt] = load_b_frag(&Bbuf[cur][(wn * 64 + nt * 16) * 32], 32, 0);
#pragma unroll
        for (int mt = 0; mt < 2; ++mt)
#pragma unroll
            for (int nt = 0; nt < 4; ++nt)
                acc[mt][nt] = wmma_f16(a[mt], b[nt], acc[mt][nt]);
        cur ^= 1;
    }

#pragma unroll
    for (int mt = 0; mt < 2; ++mt) {
#pragma unroll
        for (int nt = 0; nt < 4; ++nt) {
            int  col  = (int)nB + wn * 64 + nt * 16 + (lane & 15);
            long row0 = mB + wm * 32 + mt * 16 + ((lane >> 4) << 3);
            float bia = bias[col];
#pragma unroll
            for (int r = 0; r < 8; ++r) {
                long rr  = row0 + r;
                long idx = rr * N + col;
                float v  = acc[mt][nt][r] + bia;
                if constexpr (MODE == 0) {
                    out16[idx] = (_Float16)v;
                } else if constexpr (MODE == 1) {
                    long bb = rr / VTT, t = rr % VTT;   // VTT is power of 2
                    int  h  = col >> 6, dk = col & 63;
                    outVt[((bb * H_ + h) * (long)DK_ + dk) * VTT + t] = (_Float16)v;
                } else if constexpr (MODE == 2) {
                    out32[idx] = v + residual[idx];
                } else {
                    out16[idx] = (_Float16)fmaxf(v, 0.0f);
                }
            }
        }
    }
}

// ---------------------------------------------------------------------------
// Attention: one (b, h, BM query rows) per block, 8 waves.
//   Q  : (B*T_, D_) f16, head slice at col h*64
//   Kt : (B*TK, D_) f16, head slice at col h*64  (rows are keys)
//   Vt : (B*H, DK_, TK) f16  (pre-transposed V)
//   Out: (B*T_, D_) f16
// Scores kept as f16 in LDS (mask value -30000 survives f16).
// ---------------------------------------------------------------------------
template <int TK, int BM, bool MASKED>
__global__ __launch_bounds__(256) void attn_kernel(
    const _Float16* __restrict__ Q, const _Float16* __restrict__ Kt,
    const _Float16* __restrict__ Vt, _Float16* __restrict__ Out,
    const int* __restrict__ target) {
    __shared__ __align__(16) _Float16 P[BM * TK];
    int tid = threadIdx.x, lane = tid & 31, wid = tid >> 5;
    int b = blockIdx.z, h = blockIdx.y, m0 = blockIdx.x * BM;

    const _Float16* Qb = Q  + (long)b * T_ * D_ + h * DK_;
    const _Float16* Kb = Kt + (long)b * TK * D_ + h * DK_;

    // ---- Phase 1: S = (Q K^T) * 1/sqrt(DK), mask, store f16 to LDS ----
    constexpr int NTW = TK / 16 / 8;   // N-tiles per wave
#pragma unroll
    for (int nt = 0; nt < NTW; ++nt) {
        int n0 = (wid * NTW + nt) * 16;
#pragma unroll
        for (int mt = 0; mt < BM / 16; ++mt) {
            v8f acc = (v8f){};
#pragma unroll
            for (int ks = 0; ks < DK_; ks += 32) {
                v16h a = load_a_frag(Qb + (long)(m0 + mt * 16) * D_, D_, ks);
                v16h k = load_b_frag(Kb + (long)n0 * D_, D_, ks);
                acc = wmma_f16(a, k, acc);
            }
            int col = n0 + (lane & 15);
            int rl0 = mt * 16 + ((lane >> 4) << 3);
#pragma unroll
            for (int r = 0; r < 8; ++r) {
                int   rl = rl0 + r;
                float s  = acc[r] * 0.125f;   // 1/sqrt(64)
                if (MASKED) {
                    int  tq = m0 + rl;
                    bool ok = (col <= tq) && (target[b * T_ + tq] != 0);
                    if (!ok) s = MASKNEG;
                }
                P[rl * TK + col] = (_Float16)s;
            }
        }
    }
    __syncthreads();

    // ---- Phase 2: softmax over TK, rows split across waves ----
    constexpr int RPW = BM / 8;
#pragma unroll
    for (int rr = 0; rr < RPW; ++rr) {
        int row = wid * RPW + rr;
        float m = -3.4e38f;
        for (int c = lane; c < TK; c += 32) m = fmaxf(m, (float)P[row * TK + c]);
        for (int o = 16; o > 0; o >>= 1) m = fmaxf(m, __shfl_xor(m, o, 32));
        float sum = 0.0f;
        for (int c = lane; c < TK; c += 32) sum += __expf((float)P[row * TK + c] - m);
        for (int o = 16; o > 0; o >>= 1) sum += __shfl_xor(sum, o, 32);
        float inv = 1.0f / sum;
        for (int c = lane; c < TK; c += 32)
            P[row * TK + c] = (_Float16)(__expf((float)P[row * TK + c] - m) * inv);
    }
    __syncthreads();

    // ---- Phase 3: O = P @ V  (A from LDS, B from transposed V) ----
    constexpr int OT = (BM / 16) * 4;
    if (wid < OT) {
        int mt = wid >> 2, nt = wid & 3;
        const _Float16* Vb = Vt + ((long)(b * H_ + h) * DK_ + nt * 16) * TK;
        v8f acc = (v8f){};
        for (int ks = 0; ks < TK; ks += 32) {
            v16h a = load_a_frag(&P[mt * 16 * TK], TK, ks);
            v16h v = load_b_frag(Vb, TK, ks);
            acc = wmma_f16(a, v, acc);
        }
        int col = h * DK_ + nt * 16 + (lane & 15);
        int rl0 = mt * 16 + ((lane >> 4) << 3);
#pragma unroll
        for (int r = 0; r < 8; ++r)
            Out[((long)(b * T_ + m0 + rl0 + r)) * D_ + col] = (_Float16)acc[r];
    }
}

// ---------------------------------------------------------------------------
// Host orchestration
// ---------------------------------------------------------------------------
extern "C" void kernel_launch(void* const* d_in, const int* in_sizes, int n_in,
                              void* d_out, int out_size, void* d_ws, size_t ws_size,
                              hipStream_t stream) {
    (void)in_sizes; (void)n_in; (void)out_size; (void)ws_size;
    // setup_inputs dict order
    const int*   target = (const int*)  d_in[0];
    const float* memory = (const float*)d_in[1];
    const float* emb    = (const float*)d_in[2];
    const float* ln_g   = (const float*)d_in[3];
    const float* ln_b   = (const float*)d_in[4];
    const float* ff_w1  = (const float*)d_in[5];
    const float* ff_b1  = (const float*)d_in[6];
    const float* ff_w2  = (const float*)d_in[7];
    const float* ff_b2  = (const float*)d_in[8];
    const float* sa_wq = (const float*)d_in[9],  *sa_bq = (const float*)d_in[10];
    const float* ca_wq = (const float*)d_in[11], *ca_bq = (const float*)d_in[12];
    const float* sa_wk = (const float*)d_in[13], *sa_bk = (const float*)d_in[14];
    const float* ca_wk = (const float*)d_in[15], *ca_bk = (const float*)d_in[16];
    const float* sa_wv = (const float*)d_in[17], *sa_bv = (const float*)d_in[18];
    const float* ca_wv = (const float*)d_in[19], *ca_bv = (const float*)d_in[20];
    const float* sa_wo = (const float*)d_in[21], *sa_bo = (const float*)d_in[22];
    const float* ca_wo = (const float*)d_in[23], *ca_bo = (const float*)d_in[24];

    const int  M  = B_ * T_;     // 8192 decoder rows
    const int  Mm = B_ * TKM_;   // 32768 encoder rows

    // ---- workspace layout ----
    char* ws = (char*)d_ws;
    size_t off = 0;
    auto alloc_h = [&](size_t n) { _Float16* p = (_Float16*)(ws + off); off += n * 2; return p; };
    auto alloc_f = [&](size_t n) { float*    p = (float*)   (ws + off); off += n * 4; return p; };

    _Float16* wt_sa_q = alloc_h((size_t)D_ * D_);
    _Float16* wt_sa_k = alloc_h((size_t)D_ * D_);
    _Float16* wt_sa_v = alloc_h((size_t)D_ * D_);
    _Float16* wt_sa_o = alloc_h((size_t)D_ * D_);
    _Float16* wt_ca_q = alloc_h((size_t)D_ * D_);
    _Float16* wt_ca_k = alloc_h((size_t)D_ * D_);
    _Float16* wt_ca_v = alloc_h((size_t)D_ * D_);
    _Float16* wt_ca_o = alloc_h((size_t)D_ * D_);
    _Float16* wt_ff1  = alloc_h((size_t)D_ * FF_);
    _Float16* wt_ff2  = alloc_h((size_t)FF_ * D_);
    _Float16* mem16   = alloc_h((size_t)Mm * D_);
    _Float16* kmem    = alloc_h((size_t)Mm * D_);
    _Float16* vtmem   = alloc_h((size_t)Mm * D_);
    float*    X       = alloc_f((size_t)M * D_);
    _Float16* N16     = alloc_h((size_t)M * D_);
    _Float16* Q16     = alloc_h((size_t)M * D_);   // FF1 buffer aliases Q..attn
    _Float16* K16     = alloc_h((size_t)M * D_);
    _Float16* Vt16    = alloc_h((size_t)M * D_);
    _Float16* attn16  = alloc_h((size_t)M * D_);
    _Float16* FF1buf  = Q16;                       // 8192x2048 reuses 4x 8MB slots

    // ---- setup: embedding + PE, converts, encoder K/V (hoisted) ----
    embed_pe_kernel<<<M, 256, 0, stream>>>(target, emb, X);
    f2h_kernel<<<(unsigned)(((size_t)Mm * D_) / 256), 256, 0, stream>>>(memory, mem16, (long)Mm * D_);

    wt_convert_kernel<<<(D_ * D_) / 256, 256, 0, stream>>>(sa_wq, wt_sa_q, D_, D_);
    wt_convert_kernel<<<(D_ * D_) / 256, 256, 0, stream>>>(sa_wk, wt_sa_k, D_, D_);
    wt_convert_kernel<<<(D_ * D_) / 256, 256, 0, stream>>>(sa_wv, wt_sa_v, D_, D_);
    wt_convert_kernel<<<(D_ * D_) / 256, 256, 0, stream>>>(sa_wo, wt_sa_o, D_, D_);
    wt_convert_kernel<<<(D_ * D_) / 256, 256, 0, stream>>>(ca_wq, wt_ca_q, D_, D_);
    wt_convert_kernel<<<(D_ * D_) / 256, 256, 0, stream>>>(ca_wk, wt_ca_k, D_, D_);
    wt_convert_kernel<<<(D_ * D_) / 256, 256, 0, stream>>>(ca_wv, wt_ca_v, D_, D_);
    wt_convert_kernel<<<(D_ * D_) / 256, 256, 0, stream>>>(ca_wo, wt_ca_o, D_, D_);
    wt_convert_kernel<<<(D_ * FF_) / 256, 256, 0, stream>>>(ff_w1, wt_ff1, D_, FF_);
    wt_convert_kernel<<<(FF_ * D_) / 256, 256, 0, stream>>>(ff_w2, wt_ff2, FF_, D_);

    // Encoder-side K/V computed once (shared weights, constant memory input)
    gemm_f16_kernel<0, 1><<<dim3(D_ / 128, Mm / 128), 256, 0, stream>>>(
        mem16, wt_ca_k, ca_bk, nullptr, nullptr, kmem, nullptr, Mm, D_, D_);
    gemm_f16_kernel<1, TKM_><<<dim3(D_ / 128, Mm / 128), 256, 0, stream>>>(
        mem16, wt_ca_v, ca_bv, nullptr, nullptr, nullptr, vtmem, Mm, D_, D_);

    // ---- decoder stacks (shared parameters) ----
    dim3 gP(D_ / 128, M / 128);     // 512-wide projections
    dim3 gF1(FF_ / 128, M / 128);   // FFN up
    for (int s = 0; s < 3; ++s) {
        // self-attention
        ln_kernel<<<M, 256, 0, stream>>>(X, ln_g, ln_b, N16, nullptr);
        gemm_f16_kernel<0, 1><<<gP, 256, 0, stream>>>(N16, wt_sa_q, sa_bq, nullptr, nullptr, Q16, nullptr, M, D_, D_);
        gemm_f16_kernel<0, 1><<<gP, 256, 0, stream>>>(N16, wt_sa_k, sa_bk, nullptr, nullptr, K16, nullptr, M, D_, D_);
        gemm_f16_kernel<1, T_><<<gP, 256, 0, stream>>>(N16, wt_sa_v, sa_bv, nullptr, nullptr, nullptr, Vt16, M, D_, D_);
        attn_kernel<T_, 32, true><<<dim3(T_ / 32, H_, B_), 256, 0, stream>>>(
            Q16, K16, Vt16, attn16, target);
        gemm_f16_kernel<2, 1><<<gP, 256, 0, stream>>>(attn16, wt_sa_o, sa_bo, X, X, nullptr, nullptr, M, D_, D_);

        // cross-attention
        ln_kernel<<<M, 256, 0, stream>>>(X, ln_g, ln_b, N16, nullptr);
        gemm_f16_kernel<0, 1><<<gP, 256, 0, stream>>>(N16, wt_ca_q, ca_bq, nullptr, nullptr, Q16, nullptr, M, D_, D_);
        attn_kernel<TKM_, 16, false><<<dim3(T_ / 16, H_, B_), 256, 0, stream>>>(
            Q16, kmem, vtmem, attn16, nullptr);
        gemm_f16_kernel<2, 1><<<gP, 256, 0, stream>>>(attn16, wt_ca_o, ca_bo, X, X, nullptr, nullptr, M, D_, D_);

        // feed-forward
        ln_kernel<<<M, 256, 0, stream>>>(X, ln_g, ln_b, N16, nullptr);
        gemm_f16_kernel<3, 1><<<gF1, 256, 0, stream>>>(N16, wt_ff1, ff_b1, nullptr, nullptr, FF1buf, nullptr, M, FF_, D_);
        gemm_f16_kernel<2, 1><<<gP, 256, 0, stream>>>(FF1buf, wt_ff2, ff_b2, X, X, nullptr, nullptr, M, D_, FF_);
    }

    // ---- final LayerNorm -> f32 output ----
    ln_kernel<<<M, 256, 0, stream>>>(X, ln_g, ln_b, nullptr, (float*)d_out);
}
